// Word2ManSkipGramNegSampling_83391085019284
// MI455X (gfx1250) — matrix operand order repaired and checked by reading, not compiled, not run on hardware
//
#include <hip/hip_runtime.h>
#include <stdint.h>

typedef __attribute__((ext_vector_type(16))) __bf16 v16bf;
typedef __attribute__((ext_vector_type(8)))  float  v8f;

#define DEMB   128
#define MPITCH 136                                   // 128 + 8 pad (keeps 16B alignment, breaks bank aliasing)
#define SMEM_BYTES (2u * DEMB * MPITCH * sizeof(uint16_t))

union V16U { v16bf v; uint32_t u[8]; };

__device__ __forceinline__ uint32_t bf16rn(float a) {
    uint32_t u = __builtin_bit_cast(uint32_t, a);
    return (u + 0x7fffu + ((u >> 16) & 1u)) >> 16;   // round-to-nearest-even to bf16 bits
}

__global__ __launch_bounds__(256)
void w2m_bilinear_wmma(const int* __restrict__ xs, const int* __restrict__ ys,
                       const float* __restrict__ metric, const float* __restrict__ emb,
                       float* __restrict__ out, int nbatch)
{
    extern __shared__ uint16_t smem[];
    uint16_t* sMh = smem;                     // transposed metric, bf16 hi parts: [e][K] pitch MPITCH
    uint16_t* sMl = smem + DEMB * MPITCH;     // bf16 residual (lo) parts

    // ---- Stage metric into LDS: transpose + bf16 hi/lo split (coalesced global reads) ----
    for (int f = threadIdx.x; f < DEMB * DEMB; f += 256) {
        int d = f >> 7, e = f & (DEMB - 1);
        float m = metric[f];
        uint32_t u  = __builtin_bit_cast(uint32_t, m);
        uint32_t hb = u & 0xffff0000u;
        float lf = m - __builtin_bit_cast(float, hb);
        sMh[e * MPITCH + d] = (uint16_t)(hb >> 16);
        sMl[e * MPITCH + d] = (uint16_t)bf16rn(lf);
    }
    __syncthreads();

    const int lane = threadIdx.x & 31;
    const int l    = lane & 15;      // row-in-tile (A/M) or column-in-tile (B/N)
    const int kh   = lane >> 4;      // which K-half this lane carries
    const int wave = threadIdx.x >> 5;
    const int base = blockIdx.x * 128 + wave * 16;

    const int xrow = xs[min(base + l, nbatch - 1)];
    const float* xp = emb + (size_t)xrow * DEMB;

    v8f acc[8] = {};                 // 8 N-tiles of 16x16 f32 accumulators (T = Xe @ M)

    #pragma unroll
    for (int kk = 0; kk < 4; ++kk) {             // K step of 32
        // This lane's two contiguous 8-float A chunks (per ISA 16-bit A 16x32 layout).
        float cf[16];
        const float4* p0 = (const float4*)(xp + kk * 32 + kh * 8);
        const float4* p1 = (const float4*)(xp + kk * 32 + 16 + kh * 8);
        float4 a0 = p0[0], a1 = p0[1], b0 = p1[0], b1 = p1[1];
        cf[0]=a0.x;  cf[1]=a0.y;  cf[2]=a0.z;  cf[3]=a0.w;
        cf[4]=a1.x;  cf[5]=a1.y;  cf[6]=a1.z;  cf[7]=a1.w;
        cf[8]=b0.x;  cf[9]=b0.y;  cf[10]=b0.z; cf[11]=b0.w;
        cf[12]=b1.x; cf[13]=b1.y; cf[14]=b1.z; cf[15]=b1.w;

        V16U Ah, Al;                             // bf16 hi / residual-lo A fragments
        #pragma unroll
        for (int i = 0; i < 8; ++i) {
            int c = (i >> 2) * 8 + (i & 3) * 2;  // K_rel = (i>>2)*16 + kh*8 + 2*(i&3)
            float e0 = cf[c], e1 = cf[c + 1];
            uint32_t u0 = __builtin_bit_cast(uint32_t, e0);
            uint32_t u1 = __builtin_bit_cast(uint32_t, e1);
            Ah.u[i] = (u0 >> 16) | (u1 & 0xffff0000u);
            float l0 = e0 - __builtin_bit_cast(float, u0 & 0xffff0000u);
            float l1 = e1 - __builtin_bit_cast(float, u1 & 0xffff0000u);
            Al.u[i] = bf16rn(l0) | (bf16rn(l1) << 16);
        }

        #pragma unroll
        for (int nt = 0; nt < 8; ++nt) {         // N tile of 16
            // B fragment: K values kh*8+0..7 and 16+kh*8+0..7 of column (nt*16 + l), matching A.
            const uint16_t* bhp = sMh + (nt * 16 + l) * MPITCH + kk * 32 + kh * 8;
            const uint16_t* blp = sMl + (nt * 16 + l) * MPITCH + kk * 32 + kh * 8;
            V16U Bh, Bl;
            uint4 q0 = *(const uint4*)bhp, q1 = *(const uint4*)(bhp + 16);
            Bh.u[0]=q0.x; Bh.u[1]=q0.y; Bh.u[2]=q0.z; Bh.u[3]=q0.w;
            Bh.u[4]=q1.x; Bh.u[5]=q1.y; Bh.u[6]=q1.z; Bh.u[7]=q1.w;
            uint4 r0 = *(const uint4*)blp, r1 = *(const uint4*)(blp + 16);
            Bl.u[0]=r0.x; Bl.u[1]=r0.y; Bl.u[2]=r0.z; Bl.u[3]=r0.w;
            Bl.u[4]=r1.x; Bl.u[5]=r1.y; Bl.u[6]=r1.z; Bl.u[7]=r1.w;

            // bf16x3: lo terms first, hi*hi last (all f32-accumulated).
            acc[nt] = __builtin_amdgcn_wmma_f32_16x16x32_bf16(false, Al.v, false, Bh.v,
                                                              (short)0, acc[nt], false, false);
            acc[nt] = __builtin_amdgcn_wmma_f32_16x16x32_bf16(false, Ah.v, false, Bl.v,
                                                              (short)0, acc[nt], false, false);
            acc[nt] = __builtin_amdgcn_wmma_f32_16x16x32_bf16(false, Ah.v, false, Bh.v,
                                                              (short)0, acc[nt], false, false);
        }
    }

    // ---- Phase 2: d[M] = dot(T[M,:], ye[M,:]) using the C layout directly ----
    // acc[nt][c] = T[M = kh*8 + c][N = nt*16 + l]
    int yr[8];
    #pragma unroll
    for (int c = 0; c < 8; ++c)
        yr[c] = ys[min(base + kh * 8 + c, nbatch - 1)];

    float p[8] = {0.f,0.f,0.f,0.f,0.f,0.f,0.f,0.f};
    #pragma unroll
    for (int nt = 0; nt < 8; ++nt) {
        #pragma unroll
        for (int c = 0; c < 8; ++c) {
            // lanes 0..15 of each half read 64B contiguous segments of ye -> coalesced
            float yv = emb[(size_t)yr[c] * DEMB + nt * 16 + l];
            p[c] += acc[nt][c] * yv;
        }
    }

    // Reduce across the 16 lanes of each half (xor masks stay inside the group).
    #pragma unroll
    for (int m = 8; m >= 1; m >>= 1) {
        #pragma unroll
        for (int c = 0; c < 8; ++c)
            p[c] += __shfl_xor(p[c], m, 32);
    }

    if (l == 0) {                                 // lanes 0 and 16 write 8 outputs each
        int ob = base + kh * 8;
        float s[8];
        #pragma unroll
        for (int c = 0; c < 8; ++c)
            s[c] = 1.0f / (1.0f + __expf(-p[c]));
        if (ob + 7 < nbatch) {
            float4 o0 = {s[0], s[1], s[2], s[3]};
            float4 o1 = {s[4], s[5], s[6], s[7]};
            *(float4*)(out + ob)     = o0;
            *(float4*)(out + ob + 4) = o1;
        } else {
            for (int c = 0; c < 8; ++c)
                if (ob + c < nbatch) out[ob + c] = s[c];
        }
    }
}

extern "C" void kernel_launch(void* const* d_in, const int* in_sizes, int n_in,
                              void* d_out, int out_size, void* d_ws, size_t ws_size,
                              hipStream_t stream) {
    const int*   xs     = (const int*)d_in[0];
    const int*   ys     = (const int*)d_in[1];
    const float* metric = (const float*)d_in[2];
    const float* emb    = (const float*)d_in[3];
    float*       out    = (float*)d_out;
    const int nbatch = in_sizes[0];

    int grid = (nbatch + 127) / 128;              // 128 batch rows per 256-thread block (8 waves x 16 rows)
    w2m_bilinear_wmma<<<grid, 256, SMEM_BYTES, stream>>>(xs, ys, metric, emb, out, nbatch);
}
